// MultilayerGRU_20572893347954
// MI455X (gfx1250) — matrix operand order repaired and checked by reading, not compiled, not run on hardware
//
#include <hip/hip_runtime.h>

// GRU dims (fixed by reference)
#define BB 32
#define SS 512
#define DD 1024
#define HH 1024
#define OO 1024
#define LL 2

typedef __attribute__((ext_vector_type(16))) __bf16 v16bf;
typedef __attribute__((ext_vector_type(8)))  __bf16 v8bf;
typedef __attribute__((ext_vector_type(8)))  float  v8f;

__device__ __forceinline__ unsigned short f2bf_bits(float f) {
    unsigned int u = __float_as_uint(f);
    u += 0x7FFFu + ((u >> 16) & 1u);          // round-to-nearest-even
    return (unsigned short)(u >> 16);
}

// Accumulate acc += A[16 x K] * W[n rows x K]^T for this wave's 16x16 tile.
// A: pre-offset to wave's m_base row; row stride a_stride (elements).
// W: pre-offset to wave's n_base row; row stride K (elements, row-major [out,in]).
// Register layouts used here (contiguous k, not the ISA table order; A and B match):
//   A (16-bit, 16x32): lane-group g holds K = {16g..16g+15}
//   B (16-bit, 32x16): lane-group g holds K = {16g..16g+15}
__device__ __forceinline__ v8f wmma_acc_k(const __bf16* __restrict__ A, long a_stride,
                                          const __bf16* __restrict__ W, int K, v8f acc) {
    const int lane = threadIdx.x & 31;
    const int c16  = lane & 15;
    const int grp  = lane >> 4;
    const __bf16* ap = A + (long)c16 * a_stride;
    const __bf16* wp = W + (long)c16 * (long)K;
    #pragma unroll 4
    for (int k = 0; k < K; k += 32) {
        v16bf a = *(const v16bf*)(ap + k + 16 * grp);
        v16bf b = *(const v16bf*)(wp + k + 16 * grp);
        acc = __builtin_amdgcn_wmma_f32_16x16x32_bf16(false, a, false, b, (short)0, acc, false, false);
    }
    return acc;
}

// ---------- per-step kernels ----------

// grid = 32 blocks x 256 thr. Block b: gate = b>>4 (0=z,1=r), ntile = b&15.
// Computes gate = sigmoid(x@Wx^T + h_prev@Wh^T + bias); z -> Z (f32),
// r -> RH = bf16(r * h_prev_f32).
__global__ void gru_zr_kernel(const unsigned short* __restrict__ xA, long x_stride,
                              const unsigned short* __restrict__ Wx_z,
                              const unsigned short* __restrict__ Wx_r, int Kx,
                              const unsigned short* __restrict__ hA,
                              const unsigned short* __restrict__ Wh_z,
                              const unsigned short* __restrict__ Wh_r,
                              const float* __restrict__ bz, const float* __restrict__ br,
                              const float* __restrict__ hprev,
                              float* __restrict__ Z, unsigned short* __restrict__ RH) {
    const int gate  = blockIdx.x >> 4;
    const int ntile = blockIdx.x & 15;
    const int w = threadIdx.x >> 5, lane = threadIdx.x & 31;
    const int m_base = (w >> 2) * 16;
    const int n_base = ntile * 64 + (w & 3) * 16;
    const __bf16* Wx = (const __bf16*)(gate ? Wx_r : Wx_z);
    const __bf16* Wh = (const __bf16*)(gate ? Wh_r : Wh_z);
    const float* bias = gate ? br : bz;

    v8f acc = {};
    acc = wmma_acc_k((const __bf16*)xA + (long)m_base * x_stride, x_stride,
                     Wx + (size_t)n_base * Kx, Kx, acc);
    acc = wmma_acc_k((const __bf16*)hA + (long)m_base * HH, HH,
                     Wh + (size_t)n_base * HH, HH, acc);

    const int grp = lane >> 4;
    const int col = n_base + (lane & 15);
    const float bcol = bias[col];
    #pragma unroll
    for (int i = 0; i < 8; ++i) {
        const int row = m_base + i + 8 * grp;
        const size_t idx = (size_t)row * HH + col;
        const float s = 1.0f / (1.0f + __expf(-(acc[i] + bcol)));
        if (gate == 0) Z[idx] = s;
        else           RH[idx] = f2bf_bits(s * hprev[idx]);
    }
}

// grid = 16 blocks x 256 thr. g = tanh(x@Wxg^T + RH@Whg^T + bg);
// h = Z*h_prev + (1-Z)*g ; writes hF (f32), hBF (bf16), optionally hseq (bf16).
__global__ void gru_gh_kernel(const unsigned short* __restrict__ xA, long x_stride,
                              const unsigned short* __restrict__ Wx_g, int Kx,
                              const unsigned short* __restrict__ RH,
                              const unsigned short* __restrict__ Wh_g,
                              const float* __restrict__ bg,
                              const float* __restrict__ Z, const float* __restrict__ hprev,
                              float* __restrict__ hF, unsigned short* __restrict__ hBF,
                              unsigned short* __restrict__ hseq) {
    const int ntile = blockIdx.x;
    const int w = threadIdx.x >> 5, lane = threadIdx.x & 31;
    const int m_base = (w >> 2) * 16;
    const int n_base = ntile * 64 + (w & 3) * 16;

    v8f acc = {};
    acc = wmma_acc_k((const __bf16*)xA + (long)m_base * x_stride, x_stride,
                     (const __bf16*)Wx_g + (size_t)n_base * Kx, Kx, acc);
    acc = wmma_acc_k((const __bf16*)RH + (long)m_base * HH, HH,
                     (const __bf16*)Wh_g + (size_t)n_base * HH, HH, acc);

    const int grp = lane >> 4;
    const int col = n_base + (lane & 15);
    const float bcol = bg[col];
    #pragma unroll
    for (int i = 0; i < 8; ++i) {
        const int row = m_base + i + 8 * grp;
        const size_t idx = (size_t)row * HH + col;
        const float g = tanhf(acc[i] + bcol);
        const float z = Z[idx];
        const float h = z * hprev[idx] + (1.0f - z) * g;
        hF[idx] = h;
        const unsigned short hb = f2bf_bits(h);
        hBF[idx] = hb;
        if (hseq) hseq[idx] = hb;
    }
}

// Final output GEMM: out[b,t,:] = h1seq[t*B+b,:] @ Why^T + by
// grid = (S*B/32) * (O/64) = 8192 blocks x 256 thr.
__global__ void gru_y_kernel(const unsigned short* __restrict__ h1seq,
                             const unsigned short* __restrict__ WhyB,
                             const float* __restrict__ by, float* __restrict__ out) {
    const int mtile = blockIdx.x >> 4;
    const int ntile = blockIdx.x & 15;
    const int w = threadIdx.x >> 5, lane = threadIdx.x & 31;
    const int m_base = mtile * 32 + (w >> 2) * 16;
    const int n_base = ntile * 64 + (w & 3) * 16;

    v8f acc = {};
    acc = wmma_acc_k((const __bf16*)h1seq + (size_t)m_base * HH, HH,
                     (const __bf16*)WhyB + (size_t)n_base * HH, HH, acc);

    const int grp = lane >> 4;
    const int col = n_base + (lane & 15);
    const float bcol = by[col];
    #pragma unroll
    for (int i = 0; i < 8; ++i) {
        const int row = m_base + i + 8 * grp;   // row = t*B + b
        const int t = row >> 5, b = row & 31;
        out[(size_t)b * SS * OO + (size_t)t * OO + col] = acc[i] + bcol;
    }
}

// ---------- utility kernels ----------

__global__ void cvt_bf16_kernel(const float* __restrict__ src,
                                unsigned short* __restrict__ dst, int n) {
    int i = blockIdx.x * blockDim.x + threadIdx.x;
    const int stride = gridDim.x * blockDim.x;
    for (; i < n; i += stride) dst[i] = f2bf_bits(src[i]);
}

// hF layout [l][b][h] <- hidden_state [b][l][h]
__global__ void init_h_kernel(const float* __restrict__ hs,
                              float* __restrict__ hF, unsigned short* __restrict__ hBF) {
    const int i = blockIdx.x * blockDim.x + threadIdx.x;   // over L*B*H
    const int h = i % HH;
    const int b = (i / HH) % BB;
    const int l = i / (BB * HH);
    const float v = hs[(size_t)b * LL * HH + (size_t)l * HH + h];
    hF[i] = v;
    hBF[i] = f2bf_bits(v);
}

// dst hidden [b][l][h] <- hF [l][b][h]
__global__ void copy_hidden_kernel(const float* __restrict__ hF, float* __restrict__ dst) {
    const int i = blockIdx.x * blockDim.x + threadIdx.x;   // over L*B*H
    const int h = i % HH;
    const int b = (i / HH) % BB;
    const int l = i / (BB * HH);
    dst[(size_t)b * LL * HH + (size_t)l * HH + h] = hF[i];
}

// ---------- host ----------

extern "C" void kernel_launch(void* const* d_in, const int* in_sizes, int n_in,
                              void* d_out, int out_size, void* d_ws, size_t ws_size,
                              hipStream_t stream) {
    (void)in_sizes; (void)n_in; (void)out_size; (void)ws_size;
    const float* input  = (const float*)d_in[0];
    const float* hstate = (const float*)d_in[1];
    const float* Wxz = (const float*)d_in[2];
    const float* Whz = (const float*)d_in[3];
    const float* bz  = (const float*)d_in[4];
    const float* Wxr = (const float*)d_in[5];
    const float* Whr = (const float*)d_in[6];
    const float* br  = (const float*)d_in[7];
    const float* Wxg = (const float*)d_in[8];
    const float* Whg = (const float*)d_in[9];
    const float* bg  = (const float*)d_in[10];
    const float* Why = (const float*)d_in[11];
    const float* by  = (const float*)d_in[12];

    float* out        = (float*)d_out;
    float* out_hidden = out + (size_t)BB * SS * OO;

    // workspace carve-up (256B aligned chunks)
    char* p = (char*)d_ws;
    auto carve = [&](size_t bytes) { char* r = p; p += (bytes + 255) & ~(size_t)255; return r; };
    const size_t WEL = (size_t)LL * HH * DD;        // elements per stacked weight tensor
    unsigned short* WxzB = (unsigned short*)carve(WEL * 2);
    unsigned short* WhzB = (unsigned short*)carve(WEL * 2);
    unsigned short* WxrB = (unsigned short*)carve(WEL * 2);
    unsigned short* WhrB = (unsigned short*)carve(WEL * 2);
    unsigned short* WxgB = (unsigned short*)carve(WEL * 2);
    unsigned short* WhgB = (unsigned short*)carve(WEL * 2);
    unsigned short* WhyB = (unsigned short*)carve((size_t)OO * HH * 2);
    unsigned short* xbf  = (unsigned short*)carve((size_t)BB * SS * DD * 2);
    unsigned short* h1sq = (unsigned short*)carve((size_t)SS * BB * HH * 2);
    float*          hF   = (float*)carve((size_t)LL * BB * HH * 4);
    unsigned short* hBF  = (unsigned short*)carve((size_t)LL * BB * HH * 2);
    float*          Zbuf = (float*)carve((size_t)BB * HH * 4);
    unsigned short* RH   = (unsigned short*)carve((size_t)BB * HH * 2);

    // precision conversions (once per call)
    const int CB = 256, CG = 2048;
    cvt_bf16_kernel<<<CG, CB, 0, stream>>>(Wxz, WxzB, (int)WEL);
    cvt_bf16_kernel<<<CG, CB, 0, stream>>>(Whz, WhzB, (int)WEL);
    cvt_bf16_kernel<<<CG, CB, 0, stream>>>(Wxr, WxrB, (int)WEL);
    cvt_bf16_kernel<<<CG, CB, 0, stream>>>(Whr, WhrB, (int)WEL);
    cvt_bf16_kernel<<<CG, CB, 0, stream>>>(Wxg, WxgB, (int)WEL);
    cvt_bf16_kernel<<<CG, CB, 0, stream>>>(Whg, WhgB, (int)WEL);
    cvt_bf16_kernel<<<CG, CB, 0, stream>>>(Why, WhyB, OO * HH);
    cvt_bf16_kernel<<<CG, CB, 0, stream>>>(input, xbf, BB * SS * DD);
    init_h_kernel<<<(LL * BB * HH) / 256, 256, 0, stream>>>(hstate, hF, hBF);

    const size_t HD = (size_t)HH * DD;    // per-layer weight stride (elements)
    const size_t BH = (size_t)BB * HH;

    for (int t = 0; t < SS; ++t) {
        const unsigned short* x0 = xbf + (size_t)t * DD;     // rows strided by S*D
        // ---- layer 0 ----
        gru_zr_kernel<<<32, 256, 0, stream>>>(
            x0, (long)SS * DD, WxzB, WxrB, DD,
            hBF, WhzB, WhrB, bz, br, hF, Zbuf, RH);
        gru_gh_kernel<<<16, 256, 0, stream>>>(
            x0, (long)SS * DD, WxgB, DD,
            RH, WhgB, bg, Zbuf, hF, hF, hBF, (unsigned short*)nullptr);
        // ---- layer 1 (x = layer-0 h at step t) ----
        gru_zr_kernel<<<32, 256, 0, stream>>>(
            hBF, (long)HH, WxzB + HD, WxrB + HD, HH,
            hBF + BH, WhzB + HD, WhrB + HD, bz + HH, br + HH,
            hF + BH, Zbuf, RH);
        gru_gh_kernel<<<16, 256, 0, stream>>>(
            hBF, (long)HH, WxgB + HD, HH,
            RH, WhgB + HD, bg + HH, Zbuf, hF + BH,
            hF + BH, hBF + BH, h1sq + (size_t)t * BH);
    }

    // y = h1seq @ Why^T + by  -> out [B,S,O]
    gru_y_kernel<<<(SS * BB / 32) * (OO / 64), 256, 0, stream>>>(h1sq, WhyB, by, out);
    copy_hidden_kernel<<<(LL * BB * HH) / 256, 256, 0, stream>>>(hF, out_hidden);
}
